// ConvHDC_29240137351863
// MI455X (gfx1250) — matrix operand order repaired and checked by reading, not compiled, not run on hardware
//
#include <hip/hip_runtime.h>
#include <hip/hip_bf16.h>

// ---------------------------------------------------------------------------
// ConvHDC fused pipeline for gfx1250 (MI455X).
//   conv1 (92% of MACs) -> v_wmma_f32_16x16x32_bf16 via im2col GEMM:
//       patches[2704 x 32](bf16)  x  sign(W1)[32 x 16](bf16, exact +-1)
//   One workgroup per 16 channels; everything after conv1 is depthwise and
//   per-channel independent, kept entirely in LDS as int8 signs.
//   BN is training-mode (batch stats), so conv biases cancel and are dropped.
//   WMMA loop is 2x unrolled for back-to-back independent XDL issue.
// ---------------------------------------------------------------------------

typedef __attribute__((ext_vector_type(16))) __bf16 v16bf;
typedef __attribute__((ext_vector_type(8)))  __bf16 v8bf;
typedef __attribute__((ext_vector_type(8)))  float  v8f;

#define D_CH   10000
#define NCLS   10
#define BATCH  16
#define M1     2704      // 16 * 13 * 13 conv1 output rows
#define MT1    169       // M1 / 16 WMMA M-tiles
#define K1     32        // 25 taps padded to 32
#define CONV2_OUT 9216   // 16 batch * 36 spatial * 16 channels per WG

// ws layout (bytes)
#define PATCHES_OFF 0         // 2704*32*2 = 173056
#define BSIGN_OFF   173056    // 10000*32*2 = 640000
#define ACC_OFF     813056    // 160 floats

__device__ __forceinline__ float slrelu(float x) {
    float y = x > 0.f ? x : 0.01f * x;   // LeakyReLU(0.01)
    return y - 0.5f;                     // shifted down by 0.5
}
__device__ __forceinline__ int isign(float x) {
    return (x > 0.f) - (x < 0.f);
}

__device__ __forceinline__ v16bf wmma_bf16(v16bf a, v16bf b, v8f& c) {
    c = __builtin_amdgcn_wmma_f32_16x16x32_bf16(false, a, false, b,
                                                (short)0, c, false, false);
    return a;
}

__device__ __forceinline__ v16bf load_afrag(const __bf16* __restrict__ patches,
                                            int mt, int nsub, int hi) {
    // 16-bit A layout: lane row = nsub, K chunks {base..base+7, base+16..base+23},
    // base = hi*8  (ISA 7.12.2)
    union { v16bf v; v8bf h[2]; } ua;
    const __bf16* arow = patches + (size_t)(mt * 16 + nsub) * K1 + hi * 8;
    ua.h[0] = *(const v8bf*)(arow);
    ua.h[1] = *(const v8bf*)(arow + 16);
    return ua.v;
}

// ---------------------------------------------------------------------------
// Prep: zero the output accumulator, build bf16 im2col patches of x, and
// build the binarized-W1 B-matrix rows.
// ---------------------------------------------------------------------------
__global__ void hdc_prep(const float* __restrict__ x, const float* __restrict__ W1,
                         __bf16* __restrict__ patches, __bf16* __restrict__ bsign,
                         float* __restrict__ acc)
{
    int idx = blockIdx.x * blockDim.x + threadIdx.x;
    if (idx < BATCH * NCLS) acc[idx] = 0.f;

    if (idx < M1 * K1) {                       // im2col: stride 2, pad 1, 5x5
        int m = idx >> 5, k = idx & 31;
        float v = 0.f;
        if (k < 25) {
            int b = m / 169, r = m % 169;
            int oy = r / 13, ox = r % 13;
            int ky = k / 5,  kx = k % 5;
            int iy = 2 * oy - 1 + ky, ix = 2 * ox - 1 + kx;
            if (iy >= 0 && iy < 28 && ix >= 0 && ix < 28)
                v = x[(b * 28 + iy) * 28 + ix];
        }
        patches[idx] = (__bf16)v;
    }
    if (idx < D_CH * K1) {                     // sign(W1), zero-padded K
        int ch = idx >> 5, k = idx & 31;
        float v = 0.f;
        if (k < 25) v = (float)isign(W1[ch * 25 + k]);
        bsign[idx] = (__bf16)v;
    }
}

// ---------------------------------------------------------------------------
// Main fused kernel: one workgroup = 16 channels, 256 threads = 8 waves.
// ---------------------------------------------------------------------------
__global__ __launch_bounds__(256) void hdc_main(
    const __bf16* __restrict__ patches, const __bf16* __restrict__ bsign,
    const float* __restrict__ g1, const float* __restrict__ be1,
    const float* __restrict__ W2, const float* __restrict__ g2, const float* __restrict__ be2,
    const float* __restrict__ W3, const float* __restrict__ g3, const float* __restrict__ be3,
    const float* __restrict__ Wc, float* __restrict__ acc)
{
    __shared__ __align__(16) signed char s1[16 * M1];   // 43264 B: signs [n][m]
    __shared__ signed char s2[CONV2_OUT];               //  9216 B: conv2 raw/signs
    __shared__ signed char w2s[16 * 9];                 // sign(W2)
    __shared__ float       w3s[16 * 36];                // fp32 W3 taps
    __shared__ float sum1[16], sq1[16], mean1[16], rs1[16];
    __shared__ float sum2[16], sq2[16], mean2[16], rs2[16];
    __shared__ float h3s[16 * 16];                      // conv3 raw [n][b]
    __shared__ float mean3[16], rs3[16];
    __shared__ signed char s3[16 * 16];                 // final signs [b][n]
    __shared__ float g1s[16], be1s[16], g2s[16], be2s[16], g3s[16], be3s[16];

    const int tid  = threadIdx.x;
    const int lane = tid & 31;        // gfx1250 is wave32-only
    const int wave = tid >> 5;        // 8 waves
    const int nsub = lane & 15;       // N / channel slot within the 16-ch tile
    const int hi   = lane >> 4;       // lane half selects K sub-range
    const int ch0  = blockIdx.x * 16;

    if (tid < 16) {
        g1s[tid] = g1[ch0 + tid];  be1s[tid] = be1[ch0 + tid];
        g2s[tid] = g2[ch0 + tid];  be2s[tid] = be2[ch0 + tid];
        g3s[tid] = g3[ch0 + tid];  be3s[tid] = be3[ch0 + tid];
        sum1[tid] = 0.f; sq1[tid] = 0.f; sum2[tid] = 0.f; sq2[tid] = 0.f;
    }
    if (tid < 144) {
        int n = tid / 9, k = tid % 9;
        w2s[tid] = (signed char)isign(W2[(ch0 + n) * 9 + k]);
    }
    for (int i = tid; i < 16 * 36; i += 256)
        w3s[i] = W3[ch0 * 36 + i];    // [n][p], contiguous per channel

    // B fragment: column nsub, K chunks {hi*8..hi*8+7, 16+hi*8..16+hi*8+7} (matches A).
    v16bf bfrag;
    {
        union { v16bf v; v8bf h[2]; } ub;
        const __bf16* brow = bsign + (size_t)(ch0 + nsub) * K1 + hi * 8;
        ub.h[0] = *(const v8bf*)(brow);
        ub.h[1] = *(const v8bf*)(brow + 16);
        bfrag = ub.v;
    }

    __syncthreads();

    // ---- conv1 pass 1: WMMA (2x unrolled), accumulate BN statistics ----
    float lsum = 0.f, lsq = 0.f;
    {
        int mt = wave * 2;
        for (; mt + 1 < MT1; mt += 16) {        // pairs of independent tiles
            v16bf a0 = load_afrag(patches, mt,     nsub, hi);
            v16bf a1 = load_afrag(patches, mt + 1, nsub, hi);
            v8f c0 = {}, c1 = {};
            wmma_bf16(a0, bfrag, c0);
            wmma_bf16(a1, bfrag, c1);
#pragma unroll
            for (int i = 0; i < 8; ++i) {
                float u = c0[i], v = c1[i];
                lsum += u + v;  lsq += u * u + v * v;
            }
        }
        if (mt < MT1) {                          // wave-uniform odd tail (tile 168)
            v16bf a0 = load_afrag(patches, mt, nsub, hi);
            v8f c0 = {};
            wmma_bf16(a0, bfrag, c0);
#pragma unroll
            for (int i = 0; i < 8; ++i) { float u = c0[i]; lsum += u; lsq += u * u; }
        }
    }
    atomicAdd(&sum1[nsub], lsum);   // C fragment: every value in a lane is channel nsub
    atomicAdd(&sq1[nsub],  lsq);
    __syncthreads();
    if (tid < 16) {
        float m   = sum1[tid] / (float)M1;
        float var = sq1[tid] / (float)M1 - m * m;
        mean1[tid] = m;  rs1[tid] = rsqrtf(var + 1e-5f);
    }
    __syncthreads();

    // ---- conv1 pass 2: WMMA again, BN + shifted-leaky-relu + binarize ----
    {
        float mn = mean1[nsub], rr = rs1[nsub], gg = g1s[nsub], bb = be1s[nsub];
        signed char* srow = s1 + nsub * M1;      // channel-major sign row
        int mt = wave * 2;
        for (; mt + 1 < MT1; mt += 16) {
            v16bf a0 = load_afrag(patches, mt,     nsub, hi);
            v16bf a1 = load_afrag(patches, mt + 1, nsub, hi);
            v8f c0 = {}, c1 = {};
            wmma_bf16(a0, bfrag, c0);
            wmma_bf16(a1, bfrag, c1);
            unsigned w0a = 0, w1a = 0, w0b = 0, w1b = 0;
#pragma unroll
            for (int i = 0; i < 4; ++i) {
                unsigned sa = (unsigned char)(signed char)isign(slrelu((c0[i    ] - mn) * rr * gg + bb));
                unsigned sb = (unsigned char)(signed char)isign(slrelu((c0[i + 4] - mn) * rr * gg + bb));
                unsigned sc = (unsigned char)(signed char)isign(slrelu((c1[i    ] - mn) * rr * gg + bb));
                unsigned sd = (unsigned char)(signed char)isign(slrelu((c1[i + 4] - mn) * rr * gg + bb));
                w0a |= sa << (8 * i);  w1a |= sb << (8 * i);
                w0b |= sc << (8 * i);  w1b |= sd << (8 * i);
            }
            // 8 contiguous sign bytes per lane -> one 8B-aligned b64 store each
            *(uint2*)(srow + mt * 16 + hi * 8)        = make_uint2(w0a, w1a);
            *(uint2*)(srow + (mt + 1) * 16 + hi * 8)  = make_uint2(w0b, w1b);
        }
        if (mt < MT1) {
            v16bf a0 = load_afrag(patches, mt, nsub, hi);
            v8f c0 = {};
            wmma_bf16(a0, bfrag, c0);
            unsigned w0a = 0, w1a = 0;
#pragma unroll
            for (int i = 0; i < 4; ++i) {
                unsigned sa = (unsigned char)(signed char)isign(slrelu((c0[i    ] - mn) * rr * gg + bb));
                unsigned sb = (unsigned char)(signed char)isign(slrelu((c0[i + 4] - mn) * rr * gg + bb));
                w0a |= sa << (8 * i);  w1a |= sb << (8 * i);
            }
            *(uint2*)(srow + mt * 16 + hi * 8) = make_uint2(w0a, w1a);
        }
    }
    __syncthreads();

    // ---- conv2: depthwise 3x3 stride 2 on +-1 data (integer accumulate) ----
    const int n2 = tid & 15;   // stride 256 keeps low 4 bits fixed per thread
    const signed char* s1row = s1 + n2 * M1;
    float l2sum = 0.f, l2sq = 0.f;
    for (int o = tid; o < CONV2_OUT; o += 256) {
        int rest = o >> 4;                 // 0..575 = b*36 + p
        int b = rest / 36, p = rest % 36;
        int oy = p / 6,    ox = p % 6;
        int a = 0;
#pragma unroll
        for (int ky = 0; ky < 3; ++ky)
#pragma unroll
            for (int kx = 0; kx < 3; ++kx) {
                int iy = oy * 2 + ky, ix = ox * 2 + kx;
                a += (int)s1row[b * 169 + iy * 13 + ix]
                   * (int)w2s[n2 * 9 + ky * 3 + kx];
            }
        s2[o] = (signed char)a;            // |a| <= 9, fits int8
        float fa = (float)a;
        l2sum += fa;  l2sq += fa * fa;
    }
    atomicAdd(&sum2[n2], l2sum);
    atomicAdd(&sq2[n2],  l2sq);
    __syncthreads();
    if (tid < 16) {
        float m   = sum2[tid] / 576.f;
        float var = sq2[tid] / 576.f - m * m;
        mean2[tid] = m;  rs2[tid] = rsqrtf(var + 1e-5f);
    }
    __syncthreads();
    {
        float mn = mean2[n2], rr = rs2[n2], gg = g2s[n2], bb = be2s[n2];
        for (int o = tid; o < CONV2_OUT; o += 256) {
            float z = slrelu(((float)s2[o] - mn) * rr * gg + bb);
            s2[o] = (signed char)isign(z);
        }
    }
    __syncthreads();

    // ---- conv3: depthwise 6x6 (full spatial collapse), fp32 weights ----
    {
        int n = tid & 15, b = tid >> 4;    // 256 outputs, one per thread
        float a3 = 0.f;
#pragma unroll
        for (int p = 0; p < 36; ++p)
            a3 += (float)s2[(b * 36 + p) * 16 + n] * w3s[n * 36 + p];
        h3s[n * 16 + b] = a3;
    }
    __syncthreads();
    if (tid < 16) {
        float s = 0.f, q = 0.f;
        for (int b = 0; b < 16; ++b) { float v = h3s[tid * 16 + b]; s += v; q += v * v; }
        float m = s / 16.f;
        mean3[tid] = m;  rs3[tid] = rsqrtf(q / 16.f - m * m + 1e-5f);
    }
    __syncthreads();
    {
        int n = tid & 15, b = tid >> 4;
        float z = slrelu((h3s[n * 16 + b] - mean3[n]) * rs3[n] * g3s[n] + be3s[n]);
        s3[b * 16 + n] = (signed char)isign(z);
    }
    __syncthreads();

    // ---- classifier partial: sims[b][c] += sum_n s3 * sign(Wc) ----
    if (tid < BATCH * NCLS) {
        int b = tid / NCLS, c = tid % NCLS;
        int s = 0;
#pragma unroll
        for (int n = 0; n < 16; ++n)
            s += (int)s3[b * 16 + n] * isign(Wc[(size_t)c * D_CH + ch0 + n]);
        atomicAdd(&acc[tid], (float)s);
    }
}

__global__ void hdc_finish(const float* __restrict__ acc, float* __restrict__ out)
{
    int i = blockIdx.x * blockDim.x + threadIdx.x;
    if (i < BATCH * NCLS) out[i] = acc[i] * 0.01f;   // 1/sqrt(10000)
}

// ---------------------------------------------------------------------------
extern "C" void kernel_launch(void* const* d_in, const int* in_sizes, int n_in,
                              void* d_out, int out_size, void* d_ws, size_t ws_size,
                              hipStream_t stream)
{
    const float* x   = (const float*)d_in[0];
    const float* W1  = (const float*)d_in[1];
    // b1/b2/b3 (d_in[2], d_in[6], d_in[10]) cancel under batch-stat BN.
    const float* g1  = (const float*)d_in[3];
    const float* be1 = (const float*)d_in[4];
    const float* W2  = (const float*)d_in[5];
    const float* g2  = (const float*)d_in[7];
    const float* be2 = (const float*)d_in[8];
    const float* W3  = (const float*)d_in[9];
    const float* g3  = (const float*)d_in[11];
    const float* be3 = (const float*)d_in[12];
    const float* Wc  = (const float*)d_in[13];

    char* ws = (char*)d_ws;                     // needs ~814 KB of scratch
    __bf16* patches = (__bf16*)(ws + PATCHES_OFF);
    __bf16* bsign   = (__bf16*)(ws + BSIGN_OFF);
    float*  acc     = (float*)(ws + ACC_OFF);

    // 10000*32 = 320000 elements is the widest prep task.
    hdc_prep<<<(D_CH * K1 + 255) / 256, 256, 0, stream>>>(x, W1, patches, bsign, acc);
    hdc_main<<<D_CH / 16, 256, 0, stream>>>(patches, bsign, g1, be1,
                                            W2, g2, be2, W3, g3, be3, Wc, acc);
    hdc_finish<<<1, 256, 0, stream>>>(acc, (float*)d_out);
}
